// MoEEncoderLayer_13322988552528
// MI455X (gfx1250) — compile-verified, matrix operand order checked by reading, not run on hardware
//
#include <hip/hip_runtime.h>
#include <hip/hip_bf16.h>
#include <cstdint>
#include <cstddef>

// ---- problem constants (match reference) ----
constexpr int kD  = 1024;          // d_model
constexpr int kH  = 16;            // heads
constexpr int kHD = 64;            // head dim
constexpr int kFF = 4096;          // ffn dim
constexpr int kE  = 8;             // experts
constexpr int kL  = 2048;          // seq len
constexpr int kB  = 2;             // batch
constexpr int kT  = kL * kB;       // tokens = 4096
constexpr int k3D = 3 * kD;        // 3072

typedef __attribute__((ext_vector_type(16))) _Float16 v16h;
typedef __attribute__((ext_vector_type(8)))  float    v8f;

__device__ __forceinline__ float gelu_exact(float x) {
  return 0.5f * x * (1.0f + erff(x * 0.70710678118654752f));
}

// Async global->LDS copy of 16 bytes (gfx1250 GLOBAL_LOAD_ASYNC_TO_LDS_B128).
// ldsAddr: per-lane LDS byte address (low 32 bits of generic shared pointer);
// gaddr:   per-lane 64-bit global address. Tracked by ASYNCcnt.
__device__ __forceinline__ void async_copy_b128(unsigned ldsAddr, const void* gaddr) {
  asm volatile("global_load_async_to_lds_b128 %0, %1, off"
               :
               : "v"(ldsAddr), "v"((unsigned long long)(uintptr_t)gaddr)
               : "memory");
}
__device__ __forceinline__ void wait_asynccnt0() {
  asm volatile("s_wait_asynccnt 0x0" ::: "memory");
}

// -------------------- utility kernels --------------------
__global__ void k_f32_to_f16(const float* __restrict__ in, _Float16* __restrict__ out, size_t n) {
  size_t i = (size_t)blockIdx.x * blockDim.x + threadIdx.x;
  if (i < n) out[i] = (_Float16)in[i];
}

__global__ void k_zero_f32(float* __restrict__ p, size_t n) {
  size_t i = (size_t)blockIdx.x * blockDim.x + threadIdx.x;
  if (i < n) p[i] = 0.0f;
}

// vT[b,h,c,l] = qkv[(l*B+b), 2D + h*64 + c]  (V transposed for P@V WMMA B-operand)
__global__ void k_build_vT(const _Float16* __restrict__ qkv, _Float16* __restrict__ vT) {
  unsigned i = blockIdx.x * 256u + threadIdx.x;
  if (i >= (unsigned)(kB * kH * kHD * kL)) return;
  int l = i % kL;
  int c = (i / kL) % kHD;
  int h = (i / (kL * kHD)) % kH;
  int b = i / (kL * kHD * kH);
  vT[i] = qkv[(size_t)(l * kB + b) * k3D + 2 * kD + h * kHD + c];
}

// -------------------- WMMA GEMM: C[M,N] = epi(A[M,K] * B[N,K]^T + bias) --------------------
// Block tile 128x64, 8 waves, each wave a 32x32 macro-tile (2x2 wmma 16x16 tiles).
// LDS staging uses gfx1250 async global->LDS loads (ASYNCcnt), fragments via ds_load_b128,
// math via v_wmma_f32_16x16x32_f16 with fp32 accumulation.
// EPI: 0 = none, 1 = exact GELU.  OUT_F16: write f16 else f32.  ACC: f32 accumulate (+=).
template<int EPI, bool OUT_F16, bool ACC>
__global__ __launch_bounds__(256) void k_gemm(
    const _Float16* __restrict__ A, int lda,
    const _Float16* __restrict__ B, int ldb,
    const float* __restrict__ bias,               // len N or null
    const float* __restrict__ resid,              // f32, [M,ldc] indexing, or null
    const float* __restrict__ scale, int sstride, // per-row scale (scale[m*sstride]) or null
    void* __restrict__ Cout, int ldc,
    int M, int N, int K)
{
  (void)M; (void)N;
  __shared__ alignas(16) _Float16 As[128 * 32];
  __shared__ alignas(16) _Float16 Bs[64 * 32];

  const int tid  = threadIdx.x;
  const int lane = tid & 31;
  const int wave = tid >> 5;
  const int wm   = wave >> 1;        // 0..3 -> M sub-block
  const int wn   = wave & 1;         // 0..1 -> N sub-block
  const int mBlk = blockIdx.y * 128;
  const int nBlk = blockIdx.x * 64;
  const int lr   = tid >> 2;         // 0..63 : staging row
  const int lc   = (tid & 3) * 8;    // 0,8,16,24 : staging col (8 halves = 16B)
  const int sub  = lane >> 4;        // half-wave select for K ranges
  const int rr   = lane & 15;

  // per-lane LDS byte addresses for async staging destinations
  const unsigned ldsA0 = (unsigned)(uintptr_t)&As[(lr     ) * 32 + lc];
  const unsigned ldsA1 = (unsigned)(uintptr_t)&As[(lr + 64) * 32 + lc];
  const unsigned ldsB0 = (unsigned)(uintptr_t)&Bs[(lr     ) * 32 + lc];

  v8f acc[2][2] = {};

  for (int k0 = 0; k0 < K; k0 += 32) {
    // async stage A (128x32) and B (64x32) tiles into LDS, 16B per lane per op
    async_copy_b128(ldsA0, &A[(size_t)(mBlk + lr     ) * lda + k0 + lc]);
    async_copy_b128(ldsA1, &A[(size_t)(mBlk + lr + 64) * lda + k0 + lc]);
    async_copy_b128(ldsB0, &B[(size_t)(nBlk + lr     ) * ldb + k0 + lc]);
    if (k0 + 32 < K) {  // hint next K-chunk toward the caches (global_prefetch)
      __builtin_prefetch(&A[(size_t)(mBlk + lr) * lda + k0 + 32 + lc], 0, 3);
      __builtin_prefetch(&B[(size_t)(nBlk + lr) * ldb + k0 + 32 + lc], 0, 3);
    }
    wait_asynccnt0();
    __syncthreads();

    // fragment loads per ISA 16-bit A layout:
    // lanes 0-15: K 0-7 & 16-23 ; lanes 16-31: K 8-15 & 24-31 (two 16B chunks)
    union Frag { uint4 u[2]; v16h h; };
    Frag fa[2], fb[2];
#pragma unroll
    for (int t = 0; t < 2; t++) {
      int ar = wm * 32 + t * 16 + rr;
      fa[t].u[0] = *(const uint4*)&As[ar * 32 + sub * 8];
      fa[t].u[1] = *(const uint4*)&As[ar * 32 + sub * 8 + 16];
      int br = wn * 32 + t * 16 + rr;
      fb[t].u[0] = *(const uint4*)&Bs[br * 32 + sub * 8];
      fb[t].u[1] = *(const uint4*)&Bs[br * 32 + sub * 8 + 16];
    }
#pragma unroll
    for (int i = 0; i < 2; i++)
#pragma unroll
      for (int j = 0; j < 2; j++)
        acc[i][j] = __builtin_amdgcn_wmma_f32_16x16x32_f16(
            false, fa[i].h, false, fb[j].h, (short)0, acc[i][j], false, false);
    __syncthreads();
  }

  // epilogue: C layout — lanes 0-15: M=r, N=lane ; lanes 16-31: M=r+8, N=lane-16
  const int rbase = (lane >= 16) ? 8 : 0;
  const int nlan  = lane & 15;
#pragma unroll
  for (int i = 0; i < 2; i++) {
#pragma unroll
    for (int j = 0; j < 2; j++) {
      int m0 = mBlk + wm * 32 + i * 16;
      int n0 = nBlk + wn * 32 + j * 16 + nlan;
      float bc = bias ? bias[n0] : 0.0f;
#pragma unroll
      for (int r = 0; r < 8; r++) {
        int m = m0 + r + rbase;
        float v = acc[i][j][r] + bc;
        if (EPI == 1) v = gelu_exact(v);
        if (scale) v *= scale[(size_t)m * sstride];
        if (resid) v += resid[(size_t)m * ldc + n0];
        size_t idx = (size_t)m * ldc + n0;
        if constexpr (OUT_F16) {
          ((_Float16*)Cout)[idx] = (_Float16)v;
        } else {
          float* C = (float*)Cout;
          if constexpr (ACC) C[idx] += v; else C[idx] = v;
        }
      }
    }
  }
}

// -------------------- row softmax (in-place, f16 storage, fp32 math) --------------------
__global__ __launch_bounds__(256) void k_softmax(_Float16* __restrict__ S, int cols, float scl) {
  __shared__ float red[256];
  const int t = threadIdx.x;
  _Float16* row = S + (size_t)blockIdx.x * cols;
  const int per = cols >> 8;           // cols multiple of 256, <= 2048
  float v[8];
  float m = -3.0e38f;
  for (int i = 0; i < per; i++) { float x = (float)row[t + (i << 8)] * scl; v[i] = x; m = fmaxf(m, x); }
  red[t] = m; __syncthreads();
  for (int s = 128; s > 0; s >>= 1) { if (t < s) red[t] = fmaxf(red[t], red[t + s]); __syncthreads(); }
  m = red[0]; __syncthreads();
  float sum = 0.0f;
  for (int i = 0; i < per; i++) { v[i] = __expf(v[i] - m); sum += v[i]; }
  red[t] = sum; __syncthreads();
  for (int s = 128; s > 0; s >>= 1) { if (t < s) red[t] += red[t + s]; __syncthreads(); }
  float inv = 1.0f / red[0];
  for (int i = 0; i < per; i++) row[t + (i << 8)] = (_Float16)(v[i] * inv);
}

// -------------------- LayerNorm(x [+ a]) -> outF (f32) [+ outH (f16)] --------------------
__global__ __launch_bounds__(256) void k_add_ln(
    const float* __restrict__ x, const float* __restrict__ a,
    const float* __restrict__ g, const float* __restrict__ bta,
    float* __restrict__ outF, _Float16* __restrict__ outH)
{
  __shared__ float red[256];
  const int t = threadIdx.x;
  const size_t rb = (size_t)blockIdx.x * kD;
  float v[4]; float s = 0.0f;
  for (int i = 0; i < 4; i++) { int c = t + (i << 8); float z = x[rb + c]; if (a) z += a[rb + c]; v[i] = z; s += z; }
  red[t] = s; __syncthreads();
  for (int st = 128; st > 0; st >>= 1) { if (t < st) red[t] += red[t + st]; __syncthreads(); }
  float mean = red[0] * (1.0f / kD); __syncthreads();
  float q = 0.0f;
  for (int i = 0; i < 4; i++) { float d = v[i] - mean; q += d * d; }
  red[t] = q; __syncthreads();
  for (int st = 128; st > 0; st >>= 1) { if (t < st) red[t] += red[t + st]; __syncthreads(); }
  float rstd = rsqrtf(red[0] * (1.0f / kD) + 1e-5f);
  for (int i = 0; i < 4; i++) {
    int c = t + (i << 8);
    float y = (v[i] - mean) * rstd * g[c] + bta[c];
    outF[rb + c] = y;
    if (outH) outH[rb + c] = (_Float16)y;
  }
}

// -------------------- gate: softmax over 8 experts, top-2, renormalize --------------------
__global__ void k_gate(const float* __restrict__ x, const float* __restrict__ gW,
                       const float* __restrict__ gb, float* __restrict__ wfull) {
  int tk = blockIdx.x * blockDim.x + threadIdx.x;
  if (tk >= kT) return;
  float lg[kE];
  for (int e = 0; e < kE; e++) lg[e] = gb[e];
  const float* xr = x + (size_t)tk * kD;
  for (int d = 0; d < kD; d++) {
    float xv = xr[d];
    for (int e = 0; e < kE; e++) lg[e] += xv * gW[e * kD + d];
  }
  float mx = lg[0];
  for (int e = 1; e < kE; e++) mx = fmaxf(mx, lg[e]);
  float p[kE], sm = 0.0f;
  for (int e = 0; e < kE; e++) { p[e] = __expf(lg[e] - mx); sm += p[e]; }
  for (int e = 0; e < kE; e++) p[e] /= sm;
  int i1 = 0;
  for (int e = 1; e < kE; e++) if (p[e] > p[i1]) i1 = e;
  int i2 = (i1 == 0) ? 1 : 0;
  for (int e = 0; e < kE; e++) if (e != i1 && p[e] > p[i2]) i2 = e;
  float den = p[i1] + p[i2] + 1e-9f;
  for (int e = 0; e < kE; e++)
    wfull[(size_t)tk * kE + e] = (e == i1 || e == i2) ? p[e] / den : 0.0f;
}

// -------------------- host orchestration --------------------
extern "C" void kernel_launch(void* const* d_in, const int* in_sizes, int n_in,
                              void* d_out, int out_size, void* d_ws, size_t ws_size,
                              hipStream_t stream)
{
  (void)in_sizes; (void)n_in; (void)out_size; (void)ws_size;
  const float* src   = (const float*)d_in[0];   // [L,B,D] == tokens row (l*B+b)
  const float* Wqkv  = (const float*)d_in[1];
  const float* bqkv  = (const float*)d_in[2];
  const float* Wo    = (const float*)d_in[3];
  const float* bo    = (const float*)d_in[4];
  const float* ln1_g = (const float*)d_in[5];
  const float* ln1_b = (const float*)d_in[6];
  const float* ln2_g = (const float*)d_in[7];
  const float* ln2_b = (const float*)d_in[8];
  const float* gateW = (const float*)d_in[9];
  const float* gateb = (const float*)d_in[10];
  const float* W1    = (const float*)d_in[11];
  const float* b1    = (const float*)d_in[12];
  const float* W2    = (const float*)d_in[13];
  const float* b2    = (const float*)d_in[14];
  float* out = (float*)d_out;

  // workspace carve-out (256B aligned)
  char* base = (char*)d_ws; size_t off = 0;
  auto alloc = [&](size_t bytes) -> char* {
    char* p = base + off; off = (off + bytes + 255) & ~(size_t)255; return p;
  };
  _Float16* srcH    = (_Float16*)alloc((size_t)kT * kD * 2);
  _Float16* WqkvH   = (_Float16*)alloc((size_t)k3D * kD * 2);
  _Float16* WoH     = (_Float16*)alloc((size_t)kD * kD * 2);
  _Float16* W1H     = (_Float16*)alloc((size_t)kE * kFF * kD * 2);
  _Float16* W2H     = (_Float16*)alloc((size_t)kE * kD * kFF * 2);
  _Float16* qkvH    = (_Float16*)alloc((size_t)kT * k3D * 2);
  _Float16* vT      = (_Float16*)alloc((size_t)kB * kH * kHD * kL * 2);
  _Float16* scoresH = (_Float16*)alloc((size_t)kL * kL * 2);
  _Float16* ctxH    = (_Float16*)alloc((size_t)kT * kD * 2);
  float*    sum1    = (float*)   alloc((size_t)kT * kD * 4);
  float*    src2F   = (float*)   alloc((size_t)kT * kD * 4);
  _Float16* src2H   = (_Float16*)alloc((size_t)kT * kD * 2);
  float*    wfull   = (float*)   alloc((size_t)kT * kE * 4);
  _Float16* hH      = (_Float16*)alloc((size_t)kT * kFF * 2);
  float*    moeF    = (float*)   alloc((size_t)kT * kD * 4);

  auto cvt = [&](const float* s, _Float16* d, size_t n) {
    unsigned g = (unsigned)((n + 255) / 256);
    k_f32_to_f16<<<g, 256, 0, stream>>>(s, d, n);
  };
  cvt(src,  srcH,  (size_t)kT  * kD);
  cvt(Wqkv, WqkvH, (size_t)k3D * kD);
  cvt(Wo,   WoH,   (size_t)kD  * kD);
  cvt(W1,   W1H,   (size_t)kE  * kFF * kD);
  cvt(W2,   W2H,   (size_t)kE  * kD  * kFF);

  // QKV projection: [T,3D] = src @ Wqkv^T + bqkv
  {
    dim3 g(k3D / 64, kT / 128);
    k_gemm<0, true, false><<<g, 256, 0, stream>>>(
        srcH, kD, WqkvH, kD, bqkv, nullptr, nullptr, 0, qkvH, k3D, kT, k3D, kD);
  }

  // V transposed per (b,h) for the P@V B-operand
  {
    unsigned n = (unsigned)(kB * kH * kHD * kL);
    k_build_vT<<<(n + 255) / 256, 256, 0, stream>>>(qkvH, vT);
  }

  // attention per (b,h): S = Q K^T ; softmax(S/sqrt(hd)) ; ctx = P V
  for (int b = 0; b < kB; b++) {
    for (int h = 0; h < kH; h++) {
      const _Float16* Q  = qkvH + b * k3D + h * kHD;
      const _Float16* Kp = qkvH + b * k3D + kD + h * kHD;
      {
        dim3 g(kL / 64, kL / 128);
        k_gemm<0, true, false><<<g, 256, 0, stream>>>(
            Q, kB * k3D, Kp, kB * k3D, nullptr, nullptr, nullptr, 0,
            scoresH, kL, kL, kL, kHD);
      }
      k_softmax<<<kL, 256, 0, stream>>>(scoresH, kL, 0.125f);  // 1/sqrt(64)
      const _Float16* Vt = vT + (size_t)(b * kH + h) * kHD * kL;
      {
        dim3 g(kHD / 64, kL / 128);
        k_gemm<0, true, false><<<g, 256, 0, stream>>>(
            scoresH, kL, Vt, kL, nullptr, nullptr, nullptr, 0,
            ctxH + b * kD + h * kHD, kB * kD, kL, kHD, kL);
      }
    }
  }

  // output projection + residual: sum1 = src + ctx @ Wo^T + bo
  {
    dim3 g(kD / 64, kT / 128);
    k_gemm<0, false, false><<<g, 256, 0, stream>>>(
        ctxH, kD, WoH, kD, bo, src, nullptr, 0, sum1, kD, kT, kD, kD);
  }

  // src2 = LN1(sum1)
  k_add_ln<<<kT, 256, 0, stream>>>(sum1, nullptr, ln1_g, ln1_b, src2F, src2H);

  // gate: top-2 renormalized expert weights
  k_gate<<<(kT + 255) / 256, 256, 0, stream>>>(src2F, gateW, gateb, wfull);

  // MoE FFN: dense over experts, weighted accumulate
  {
    size_t n = (size_t)kT * kD;
    k_zero_f32<<<(unsigned)((n + 255) / 256), 256, 0, stream>>>(moeF, n);
  }
  for (int e = 0; e < kE; e++) {
    {
      dim3 g(kFF / 64, kT / 128);
      k_gemm<1, true, false><<<g, 256, 0, stream>>>(
          src2H, kD, W1H + (size_t)e * kFF * kD, kD, b1 + e * kFF,
          nullptr, nullptr, 0, hH, kFF, kT, kFF, kD);
    }
    {
      dim3 g(kD / 64, kT / 128);
      k_gemm<0, false, true><<<g, 256, 0, stream>>>(
          hH, kFF, W2H + (size_t)e * kD * kFF, kFF, b2 + e * kD,
          nullptr, wfull + e, kE, moeF, kD, kT, kD, kFF);
    }
  }

  // out = LN2(src2 + moe_out)
  k_add_ln<<<kT, 256, 0, stream>>>(src2F, moeF, ln2_g, ln2_b, out, nullptr);
}
